// RGCN_26860725469622
// MI455X (gfx1250) — matrix operand order repaired and checked by reading, not compiled, not run on hardware
//
#include <hip/hip_runtime.h>
#include <hip/hip_bf16.h>
#include <stdint.h>
#include <stddef.h>

#define R_REL 65
#define HDIM  256
#define FIN1  74
#define FPAD1 96
#define NK1   3      // 96/32
#define NK2   8      // 256/32
#define MHID  64

typedef __attribute__((ext_vector_type(16))) _Float16 v16h;
typedef __attribute__((ext_vector_type(8)))  _Float16 v8h;
typedef __attribute__((ext_vector_type(8)))  float    v8f;

// ---------------------------------------------------------------------------
// WMMA fragment helpers (layouts per CDNA5 ISA §7.12.2)
// A 16x32 f16 frag: lane<16 -> row=lane, K {kb+0..7, kb+16..23}
//                   lane>=16 -> row=lane-16, K {kb+8..15, kb+24..31}
// ---------------------------------------------------------------------------
__device__ __forceinline__ v16h load_a_frag(const _Float16* Alds, int lane,
                                            int kbase, int fin_pad) {
  int m  = lane & 15;
  int ko = (lane >> 4) << 3;
  const _Float16* p = Alds + m * fin_pad + kbase + ko;
  v8h lo = *(const v8h*)(p);         // ds_load_b128
  v8h hi = *(const v8h*)(p + 16);    // ds_load_b128
  return __builtin_shufflevector(lo, hi, 0,1,2,3,4,5,6,7,8,9,10,11,12,13,14,15);
}

__device__ __forceinline__ v8f wmma_f16(v16h a, v16h b, v8f c) {
  return __builtin_amdgcn_wmma_f32_16x16x32_f16(false, a, false, b,
                                                (short)0, c, false, false);
}

// ---------------------------------------------------------------------------
// Small utility kernels
// ---------------------------------------------------------------------------
__global__ void zero_kernel(int* p, long long n) {
  long long i = (long long)blockIdx.x * blockDim.x + threadIdx.x;
  if (i < n) p[i] = 0;
}

__global__ void count_kernel(const int* __restrict__ dst,
                             const int* __restrict__ et,
                             int* __restrict__ cnt, int* __restrict__ relcnt,
                             int E) {
  int e = blockIdx.x * blockDim.x + threadIdx.x;
  if (e < E) {
    atomicAdd(&cnt[dst[e] * R_REL + et[e]], 1);
    atomicAdd(&relcnt[et[e]], 1);
  }
}

__global__ void norm_kernel(const int* __restrict__ dst,
                            const int* __restrict__ et,
                            const int* __restrict__ cnt,
                            float* __restrict__ nrm, int E) {
  int e = blockIdx.x * blockDim.x + threadIdx.x;
  if (e < E) {
    float c = (float)cnt[dst[e] * R_REL + et[e]];
    nrm[e] = 1.0f / fmaxf(c, 1.0f);
  }
}

// single-thread scan over 65 relations + tile-descriptor build
__global__ void scan_tiles_kernel(const int* __restrict__ relcnt,
                                  int* __restrict__ cursor,
                                  int* __restrict__ tiles,
                                  int* __restrict__ n_tiles) {
  int off = 0, nt = 0;
  for (int r = 0; r < R_REL; ++r) {
    int c = relcnt[r];
    cursor[r] = off;
    for (int t = 0; t < c; t += 16) {
      tiles[nt * 3 + 0] = r;
      tiles[nt * 3 + 1] = off + t;
      int rem = c - t;
      tiles[nt * 3 + 2] = rem < 16 ? rem : 16;
      ++nt;
    }
    off += c;
  }
  n_tiles[0] = nt;
}

__global__ void perm_kernel(const int* __restrict__ et, int* __restrict__ cursor,
                            int* __restrict__ perm, int E) {
  int e = blockIdx.x * blockDim.x + threadIdx.x;
  if (e < E) {
    int pos = atomicAdd(&cursor[et[e]], 1);
    perm[pos] = e;
  }
}

// Pack f32 weights [R][fin][256] into WMMA-B fragment order:
// idx = (((r*16+nt)*nk + ks)*32 + lane)*16 + h
// lane<16: N=nt*16+lane, K=ks*32+{0..7,16..23}[h] ; lane>=16: N=nt*16+lane-16, K=ks*32+{8..15,24..31}[h]
__global__ void pack_kernel(const float* __restrict__ W, _Float16* __restrict__ out,
                            int R, int fin, int nk) {
  int idx = blockIdx.x * blockDim.x + threadIdx.x;
  int total = R * 16 * nk * 512;
  if (idx >= total) return;
  int h    = idx & 15;
  int lane = (idx >> 4) & 31;
  int rest = idx >> 9;
  int ks   = rest % nk; rest /= nk;
  int nt   = rest & 15;
  int r    = rest >> 4;
  int k    = ks * 32 + ((lane >> 4) << 3) + (h & 7) + ((h >> 3) << 4);
  int col  = nt * 16 + (lane & 15);
  float v  = (k < fin) ? W[((size_t)r * fin + k) * HDIM + col] : 0.0f;
  out[idx] = (_Float16)v;
}

__global__ void relu_kernel(float* __restrict__ p, long long n) {
  long long i = (long long)blockIdx.x * blockDim.x + threadIdx.x;
  if (i < n) p[i] = fmaxf(p[i], 0.0f);
}

// ---------------------------------------------------------------------------
// Dense root GEMM: out[M,256] = f16(feat[M,fin]) @ Bpack + bias
// block = 256 threads = 8 waves; wave w owns column tiles {2w, 2w+1}
// ---------------------------------------------------------------------------
__global__ void __launch_bounds__(256) wmma_root_gemm(
    const float* __restrict__ feat, int M, int fin, int fin_pad,
    const _Float16* __restrict__ Bpack, int nk,
    const float* __restrict__ bias, float* __restrict__ out) {
  __shared__ __align__(16) _Float16 Alds[16 * HDIM];
  int tid  = threadIdx.x;
  int row0 = blockIdx.x * 16;

  for (int idx = tid; idx < 16 * fin_pad; idx += 256) {
    int r = idx / fin_pad, k = idx - r * fin_pad;
    int row = row0 + r;
    float v = (row < M && k < fin) ? feat[(size_t)row * fin + k] : 0.0f;
    Alds[r * fin_pad + k] = (_Float16)v;
  }
  __syncthreads();

  int wave = tid >> 5, lane = tid & 31;
  int nt0 = wave * 2, nt1 = nt0 + 1;
  v8f c0 = {}, c1 = {};
  for (int ks = 0; ks < nk; ++ks) {
    v16h a  = load_a_frag(Alds, lane, ks * 32, fin_pad);
    v16h b0 = *(const v16h*)(Bpack + (((size_t)nt0 * nk + ks) * 32 + lane) * 16);
    v16h b1 = *(const v16h*)(Bpack + (((size_t)nt1 * nk + ks) * 32 + lane) * 16);
    c0 = wmma_f16(a, b0, c0);
    c1 = wmma_f16(a, b1, c1);
  }
  int n = lane & 15, mbase = (lane >> 4) << 3;
#pragma unroll
  for (int v = 0; v < 8; ++v) {
    int row = row0 + mbase + v;
    if (row < M) {
      int col0 = nt0 * 16 + n, col1 = nt1 * 16 + n;
      out[(size_t)row * HDIM + col0] = c0[v] + bias[col0];
      out[(size_t)row * HDIM + col1] = c1[v] + bias[col1];
    }
  }
}

// ---------------------------------------------------------------------------
// Edge GEMM: per 16-edge relation tile, A = gathered normalized messages,
// B = packed W[rel]; scatter-add 16x256 result rows to out[dst].
// ---------------------------------------------------------------------------
__global__ void __launch_bounds__(256) wmma_edge_gemm(
    const float* __restrict__ feat, int fin, int fin_pad, int feat_stride,
    const _Float16* __restrict__ Wpack, int nk,
    const int* __restrict__ src, const int* __restrict__ dst,
    const float* __restrict__ nrm, const int* __restrict__ perm,
    const int* __restrict__ tiles, const int* __restrict__ n_tiles,
    float* __restrict__ out) {
  int tile = blockIdx.x;
  if (tile >= n_tiles[0]) return;

  __shared__ __align__(16) _Float16 Alds[16 * HDIM];
  __shared__ int s_src[16];
  __shared__ int s_dst[16];
  __shared__ float s_nrm[16];
  __shared__ int s_meta[3];

  int tid = threadIdx.x;
  if (tid < 3) s_meta[tid] = tiles[tile * 3 + tid];
  __syncthreads();
  int rel = s_meta[0], start = s_meta[1], nrows = s_meta[2];

  if (tid < 16) {
    if (tid < nrows) {
      int e = perm[start + tid];
      s_src[tid] = src[e];
      s_dst[tid] = dst[e];
      s_nrm[tid] = nrm[e];
    } else {
      s_src[tid] = 0; s_dst[tid] = -1; s_nrm[tid] = 0.0f;
    }
  }
  __syncthreads();

  for (int idx = tid; idx < 16 * fin_pad; idx += 256) {
    int r = idx / fin_pad, k = idx - r * fin_pad;
    float v = 0.0f;
    if (r < nrows && k < fin)
      v = feat[(size_t)s_src[r] * feat_stride + k] * s_nrm[r];
    Alds[r * fin_pad + k] = (_Float16)v;
  }
  __syncthreads();

  int wave = tid >> 5, lane = tid & 31;
  int nt0 = wave * 2, nt1 = nt0 + 1;
  const _Float16* Wr = Wpack + (size_t)rel * 16 * nk * 512;
  v8f c0 = {}, c1 = {};
  for (int ks = 0; ks < nk; ++ks) {
    v16h a  = load_a_frag(Alds, lane, ks * 32, fin_pad);
    v16h b0 = *(const v16h*)(Wr + (((size_t)nt0 * nk + ks) * 32 + lane) * 16);
    v16h b1 = *(const v16h*)(Wr + (((size_t)nt1 * nk + ks) * 32 + lane) * 16);
    c0 = wmma_f16(a, b0, c0);
    c1 = wmma_f16(a, b1, c1);
  }
  int n = lane & 15, mbase = (lane >> 4) << 3;
#pragma unroll
  for (int v = 0; v < 8; ++v) {
    int m = mbase + v;
    if (m < nrows) {
      float* orow = out + (size_t)s_dst[m] * HDIM;
      atomicAdd(orow + nt0 * 16 + n, c0[v]);
      atomicAdd(orow + nt1 * 16 + n, c1[v]);
    }
  }
}

// ---------------------------------------------------------------------------
// WeightedSum pooling: w = sigmoid(h·ws_w + ws_b); g[batch] += w*h
// ---------------------------------------------------------------------------
__global__ void __launch_bounds__(256) pool_kernel(
    const float* __restrict__ h, const float* __restrict__ ws_w,
    const float* __restrict__ ws_b, const int* __restrict__ batch,
    float* __restrict__ g) {
  int n = blockIdx.x, t = threadIdx.x;
  __shared__ float red[256];
  float hv = h[(size_t)n * HDIM + t];
  red[t] = hv * ws_w[t];
  __syncthreads();
  for (int s = 128; s > 0; s >>= 1) {
    if (t < s) red[t] += red[t + s];
    __syncthreads();
  }
  float w = 1.0f / (1.0f + __expf(-(red[0] + ws_b[0])));
  atomicAdd(&g[(size_t)batch[n] * HDIM + t], w * hv);
}

// ---------------------------------------------------------------------------
// MLP head
// ---------------------------------------------------------------------------
__global__ void mlp_kernel(const float* __restrict__ in, const float* __restrict__ W,
                           const float* __restrict__ b, float* __restrict__ out,
                           int K, int do_relu) {
  int i = blockIdx.x, t = threadIdx.x, O = blockDim.x;
  float acc = b[t];
  for (int k = 0; k < K; ++k) acc += in[(size_t)i * K + k] * W[k * O + t];
  if (do_relu) acc = fmaxf(acc, 0.0f);
  out[(size_t)i * O + t] = acc;
}

__global__ void mlp3_kernel(const float* __restrict__ in, const float* __restrict__ W,
                            const float* __restrict__ b, float* __restrict__ out, int G) {
  int i = blockIdx.x * blockDim.x + threadIdx.x;
  if (i < G) {
    float acc = b[0];
    for (int k = 0; k < MHID; ++k) acc += in[(size_t)i * MHID + k] * W[k];
    out[i] = acc;
  }
}

// ---------------------------------------------------------------------------
extern "C" void kernel_launch(void* const* d_in, const int* in_sizes, int n_in,
                              void* d_out, int out_size, void* d_ws, size_t ws_size,
                              hipStream_t stream) {
  const float* x     = (const float*)d_in[0];
  const int*   ei    = (const int*)d_in[1];
  const int*   et    = (const int*)d_in[2];
  const int*   batch = (const int*)d_in[3];
  const float* W1    = (const float*)d_in[4];
  const float* root1 = (const float*)d_in[5];
  const float* b1    = (const float*)d_in[6];
  const float* W2    = (const float*)d_in[7];
  const float* root2 = (const float*)d_in[8];
  const float* b2    = (const float*)d_in[9];
  const float* ws_w  = (const float*)d_in[10];
  const float* ws_b  = (const float*)d_in[11];
  const float* m1    = (const float*)d_in[12];
  const float* mb1   = (const float*)d_in[13];
  const float* m2    = (const float*)d_in[14];
  const float* mb2   = (const float*)d_in[15];
  const float* m3    = (const float*)d_in[16];
  const float* mb3   = (const float*)d_in[17];

  const int E = in_sizes[1] / 2;
  const int N = in_sizes[0] / FIN1;
  const int G = out_size;            // [G,1] output
  const int* src = ei;
  const int* dst = ei + E;

  // ---- carve workspace (256B aligned) ----
  char* p = (char*)d_ws;
  auto take = [&](size_t bytes) -> void* {
    void* q = (void*)p;
    p += (bytes + 255) & ~(size_t)255;
    return q;
  };
  int*      cnt     = (int*)take((size_t)N * R_REL * 4);
  int*      relcnt  = (int*)take(R_REL * 4);
  int*      cursor  = (int*)take(R_REL * 4);
  int*      ntile   = (int*)take(4);
  float*    nrm     = (float*)take((size_t)E * 4);
  int*      perm    = (int*)take((size_t)E * 4);
  const int maxTiles = E / 16 + R_REL + 1;
  int*      tiles   = (int*)take((size_t)maxTiles * 12);
  _Float16* W1p     = (_Float16*)take((size_t)R_REL * FPAD1 * HDIM * 2);
  _Float16* W2p     = (_Float16*)take((size_t)R_REL * HDIM * HDIM * 2);
  _Float16* r1p     = (_Float16*)take((size_t)FPAD1 * HDIM * 2);
  _Float16* r2p     = (_Float16*)take((size_t)HDIM * HDIM * 2);
  float*    h1      = (float*)take((size_t)N * HDIM * 4);
  float*    h2      = (float*)take((size_t)N * HDIM * 4);
  float*    g       = (float*)take((size_t)G * HDIM * 4);
  float*    z1      = (float*)take((size_t)G * MHID * 4);
  float*    z2      = (float*)take((size_t)G * MHID * 4);

  const int T = 256;
  auto blocks = [](long long n, int t) { return (int)((n + t - 1) / t); };

  // ---- zero accumulators (deterministic per call) ----
  zero_kernel<<<blocks((long long)N * R_REL, T), T, 0, stream>>>(cnt, (long long)N * R_REL);
  zero_kernel<<<1, T, 0, stream>>>(relcnt, R_REL);
  zero_kernel<<<blocks((long long)G * HDIM, T), T, 0, stream>>>((int*)g, (long long)G * HDIM);

  // ---- edge normalization + relation bucketing ----
  count_kernel<<<blocks(E, T), T, 0, stream>>>(dst, et, cnt, relcnt, E);
  norm_kernel<<<blocks(E, T), T, 0, stream>>>(dst, et, cnt, nrm, E);
  scan_tiles_kernel<<<1, 1, 0, stream>>>(relcnt, cursor, tiles, ntile);
  perm_kernel<<<blocks(E, T), T, 0, stream>>>(et, cursor, perm, E);

  // ---- pack weights into WMMA-B fragment order (f32 -> f16) ----
  pack_kernel<<<blocks((long long)R_REL * 16 * NK1 * 512, T), T, 0, stream>>>(W1, W1p, R_REL, FIN1, NK1);
  pack_kernel<<<blocks((long long)R_REL * 16 * NK2 * 512, T), T, 0, stream>>>(W2, W2p, R_REL, HDIM, NK2);
  pack_kernel<<<blocks((long long)16 * NK1 * 512, T), T, 0, stream>>>(root1, r1p, 1, FIN1, NK1);
  pack_kernel<<<blocks((long long)16 * NK2 * 512, T), T, 0, stream>>>(root2, r2p, 1, HDIM, NK2);

  // ---- layer 1: h1 = relu(x@root1 + b1 + scatter(msg@W1)) ----
  wmma_root_gemm<<<blocks(N, 16), T, 0, stream>>>(x, N, FIN1, FPAD1, r1p, NK1, b1, h1);
  wmma_edge_gemm<<<maxTiles, T, 0, stream>>>(x, FIN1, FPAD1, FIN1, W1p, NK1,
                                             src, dst, nrm, perm, tiles, ntile, h1);
  relu_kernel<<<blocks((long long)N * HDIM, T), T, 0, stream>>>(h1, (long long)N * HDIM);

  // ---- layer 2: h2 = relu(h1@root2 + b2 + scatter(msg@W2)) ----
  wmma_root_gemm<<<blocks(N, 16), T, 0, stream>>>(h1, N, HDIM, HDIM, r2p, NK2, b2, h2);
  wmma_edge_gemm<<<maxTiles, T, 0, stream>>>(h1, HDIM, HDIM, HDIM, W2p, NK2,
                                             src, dst, nrm, perm, tiles, ntile, h2);
  relu_kernel<<<blocks((long long)N * HDIM, T), T, 0, stream>>>(h2, (long long)N * HDIM);

  // ---- weighted-sum pooling + MLP head ----
  pool_kernel<<<N, T, 0, stream>>>(h2, ws_w, ws_b, batch, g);
  mlp_kernel<<<G, MHID, 0, stream>>>(g, m1, mb1, z1, HDIM, 1);
  mlp_kernel<<<G, MHID, 0, stream>>>(z1, m2, mb2, z2, MHID, 1);
  mlp3_kernel<<<blocks(G, T), T, 0, stream>>>(z2, m3, mb3, (float*)d_out, G);
}
